// QCNetAgentEncoder_60782377173426
// MI455X (gfx1250) — matrix operand order repaired and checked by reading, not compiled, not run on hardware
//
#include <hip/hip_runtime.h>

// ---------------------------------------------------------------------------
// QCNet agent encoder for MI455X (gfx1250, wave32, WMMA + TDM).
// GEMMs: v_wmma_f32_16x16x32_bf16, f32 accumulate. A-tiles staged to LDS via
// tensor_load_to_lds (TDM, 6-arg clang-23 form); B staged transposed in LDS
// so fragments are contiguous ds_load_b128s.
// ---------------------------------------------------------------------------

#define AA 512
#define TT 50
#define MM 1024
#define HH 128
#define NTOKENS (AA * TT)   // 25600
#define NPLS (TT * MM)      // 51200
#define PI_F 3.14159265358979323846f
#define TWO_PI_F 6.28318530717958647692f

typedef __attribute__((ext_vector_type(16))) __bf16 v16bf;
typedef __attribute__((ext_vector_type(8)))  float  v8f;
typedef __attribute__((ext_vector_type(4)))  unsigned v4u;
typedef __attribute__((ext_vector_type(8)))  int      v8i;
typedef __attribute__((ext_vector_type(4)))  int      v4i;

#if __has_builtin(__builtin_amdgcn_tensor_load_to_lds)
#define HAVE_TDM 1
#else
#define HAVE_TDM 0
#endif

// ---------------- fragment builders (ISA 7.12.2 layouts, wave32) -----------
__device__ __forceinline__ v16bf load_a_frag(const __bf16* tile, int ld, int lane) {
  // A: 16x32 bf16 row-major; per-lane halves are two contiguous 8-runs -> b128
  int m = lane & 15, hi = lane >> 4;
  v16bf a;
#pragma unroll
  for (int j = 0; j < 16; ++j) {
    int v = j >> 1, h = j & 1;
    int k = ((v >> 2) << 4) + hi * 8 + ((v & 3) << 1) + h;
    a[j] = tile[m * ld + k];
  }
  return a;
}
// B staged transposed in LDS: BsT[n*32 + k]; lane reads k = 8hi..8hi+7 and 16+8hi..16+8hi+7
__device__ __forceinline__ v16bf load_bT_frag(const __bf16* tileT, int lane) {
  int n = lane & 15, hi = lane >> 4;
  const __bf16* p = tileT + n * 32 + hi * 8;
  v16bf b;
#pragma unroll
  for (int j = 0; j < 16; ++j) b[j] = p[j < 8 ? j : j + 8];
  return b;
}

// monotonic float<->uint key for atomicMax-based segment max
__device__ __forceinline__ unsigned fkey(float f) {
  unsigned b = __float_as_uint(f);
  return (b & 0x80000000u) ? ~b : (b | 0x80000000u);
}
__device__ __forceinline__ float fdec(unsigned k) {
  unsigned b = (k & 0x80000000u) ? (k ^ 0x80000000u) : ~k;
  return __uint_as_float(b);
}
__device__ __forceinline__ float norm2f(float x, float y) { return sqrtf(x * x + y * y + 1e-12f); }
__device__ __forceinline__ float angle_btw(float cx, float cy, float nx, float ny) {
  float y = cx * ny - cy * nx;
  float x = cx * nx + cy * ny;
  if (fabsf(x) + fabsf(y) < 1e-6f) return 0.f;
  return atan2f(y, x);
}
__device__ __forceinline__ float wrapf(float a) {
  float w = a + PI_F;
  w -= floorf(w * (1.0f / TWO_PI_F)) * TWO_PI_F;
  return w - PI_F;
}

// stage a 32 x 128 bf16 weight slice (row-major, ld=128) transposed into LDS
__device__ __forceinline__ void stage_bT128(const __bf16* __restrict__ W, __bf16* BsT, int tid) {
  for (int i = tid * 8; i < 32 * 128; i += 128 * 8) {
    int r = i >> 7, c = i & 127;
#pragma unroll
    for (int j = 0; j < 8; ++j) BsT[(c + j) * 32 + r] = W[(long)r * 128 + c + j];
  }
}

// ---------------------------- utility kernels ------------------------------
__global__ void k_cvt_bf16(const float* __restrict__ s, __bf16* __restrict__ d, long n) {
  long i = (long)blockIdx.x * blockDim.x + threadIdx.x;
  if (i < n) d[i] = (__bf16)s[i];
}
__global__ void k_pad_w1(const float* __restrict__ s, __bf16* __restrict__ d, int din) {
  long i = (long)blockIdx.x * blockDim.x + threadIdx.x;
  long tot = (long)din * 160 * 128;
  if (i >= tot) return;
  int c = (int)(i % 128);
  long r = i / 128;
  int k = (int)(r % 160);
  int dd = (int)(r / 160);
  d[i] = (k < 129) ? (__bf16)s[((long)dd * 129 + k) * 128 + c] : (__bf16)0.f;
}
__global__ void k_copy(const float* __restrict__ s, float* __restrict__ d, long n) {
  long i = (long)blockIdx.x * blockDim.x + threadIdx.x;
  if (i < n) d[i] = s[i];
}
__global__ void k_perm(const float* __restrict__ s, float* __restrict__ d, int a2t) {
  long i = (long)blockIdx.x * blockDim.x + threadIdx.x;
  if (i >= (long)NTOKENS * HH) return;
  int h = (int)(i & (HH - 1));
  int n = (int)(i >> 7);
  long o;
  if (a2t) { int a = n / TT, t = n % TT; o = ((long)(t * AA + a) << 7) + h; }
  else     { int t = n / AA, a = n % AA; o = ((long)(a * TT + t) << 7) + h; }
  d[o] = s[i];
}
__global__ void k_xpls(const float* __restrict__ s, float* __restrict__ d) {
  long i = (long)blockIdx.x * blockDim.x + threadIdx.x;
  if (i >= (long)MM * TT * HH) return;
  int h = (int)(i % HH);
  long r = i / HH;
  int t = (int)(r % TT);
  int m = (int)(r / TT);
  d[((long)(t * MM + m)) * HH + h] = s[i];
}

// ---------------------------- feature kernels ------------------------------
__global__ void k_feat_agent(const float* __restrict__ pos, const float* __restrict__ head,
                             const float* __restrict__ vel, float* __restrict__ xf) {
  int n = blockIdx.x * blockDim.x + threadIdx.x;
  if (n >= NTOKENS) return;
  int t = n % TT;
  float px = pos[n * 2], py = pos[n * 2 + 1];
  float mx = 0.f, my = 0.f;
  if (t > 0) { mx = px - pos[(n - 1) * 2]; my = py - pos[(n - 1) * 2 + 1]; }
  float h = head[n], hx = cosf(h), hy = sinf(h);
  float vx = vel[n * 2], vy = vel[n * 2 + 1];
  xf[n * 4 + 0] = norm2f(mx, my);
  xf[n * 4 + 1] = angle_btw(hx, hy, mx, my);
  xf[n * 4 + 2] = norm2f(vx, vy);
  xf[n * 4 + 3] = angle_btw(hx, hy, vx, vy);
}
__global__ void k_feat_rt(const float* __restrict__ pos, const float* __restrict__ head,
                          const int* __restrict__ ei, int E, float* __restrict__ rf) {
  int e = blockIdx.x * blockDim.x + threadIdx.x;
  if (e >= E) return;
  int s = ei[e], d = ei[E + e];
  float rx = pos[s * 2] - pos[d * 2];
  float ry = pos[s * 2 + 1] - pos[d * 2 + 1];
  float hd = head[d], hx = cosf(hd), hy = sinf(hd);
  rf[e * 4 + 0] = norm2f(rx, ry);
  rf[e * 4 + 1] = angle_btw(hx, hy, rx, ry);
  rf[e * 4 + 2] = wrapf(head[s] - head[d]);
  rf[e * 4 + 3] = (float)(s - d);
}
__global__ void k_feat_rpl2a(const float* __restrict__ pos_a, const float* __restrict__ head_a,
                             const float* __restrict__ pos_pl, const float* __restrict__ orient_pl,
                             const int* __restrict__ ei, int E, float* __restrict__ rf) {
  int e = blockIdx.x * blockDim.x + threadIdx.x;
  if (e >= E) return;
  int sp = ei[e], dp = ei[E + e];
  int pl = sp % MM;
  int t = dp / AA, a = dp % AA;
  int tok = a * TT + t;
  float rx = pos_pl[pl * 2] - pos_a[tok * 2];
  float ry = pos_pl[pl * 2 + 1] - pos_a[tok * 2 + 1];
  float hd = head_a[tok], hx = cosf(hd), hy = sinf(hd);
  rf[e * 3 + 0] = norm2f(rx, ry);
  rf[e * 3 + 1] = angle_btw(hx, hy, rx, ry);
  rf[e * 3 + 2] = wrapf(orient_pl[pl] - hd);
}
__global__ void k_feat_ra2a(const float* __restrict__ pos_a, const float* __restrict__ head_a,
                            const int* __restrict__ ei, int E, float* __restrict__ rf) {
  int e = blockIdx.x * blockDim.x + threadIdx.x;
  if (e >= E) return;
  int sa = ei[e], da = ei[E + e];
  int ts = sa / AA, as_ = sa % AA, toks = as_ * TT + ts;
  int td = da / AA, ad = da % AA, tokd = ad * TT + td;
  float rx = pos_a[toks * 2] - pos_a[tokd * 2];
  float ry = pos_a[toks * 2 + 1] - pos_a[tokd * 2 + 1];
  float hd = head_a[tokd], hx = cosf(hd), hy = sinf(hd);
  rf[e * 3 + 0] = norm2f(rx, ry);
  rf[e * 3 + 1] = angle_btw(hx, hy, rx, ry);
  rf[e * 3 + 2] = wrapf(head_a[toks] - hd);
}

// ------------------------------ pe kernels ---------------------------------
__global__ void k_reduce_pe(const float* __restrict__ pos_a, const float* __restrict__ pos_pl,
                            float* __restrict__ sums) {
  __shared__ float sh[4 * 256];
  int t = threadIdx.x;
  float c0 = 0, c1 = 0, p0 = 0, p1 = 0;
  for (int a = t; a < AA; a += 256) {
    c0 += pos_a[((long)a * TT + TT - 1) * 2];
    c1 += pos_a[((long)a * TT + TT - 1) * 2 + 1];
  }
  for (int m = t; m < MM; m += 256) { p0 += pos_pl[m * 2]; p1 += pos_pl[m * 2 + 1]; }
  sh[t] = c0; sh[256 + t] = c1; sh[512 + t] = p0; sh[768 + t] = p1;
  __syncthreads();
  for (int off = 128; off; off >>= 1) {
    if (t < off)
      for (int j = 0; j < 4; ++j) sh[j * 256 + t] += sh[j * 256 + t + off];
    __syncthreads();
  }
  if (t == 0) { sums[0] = sh[0]; sums[1] = sh[256]; sums[2] = sh[512]; sums[3] = sh[768]; }
}
__global__ void k_pe(const float* __restrict__ sums, const float* __restrict__ pos_a,
                     const float* __restrict__ peW, const float* __restrict__ peb,
                     float* __restrict__ pe) {
  int i = blockIdx.x * blockDim.x + threadIdx.x;
  if (i >= AA * HH) return;
  int a = i >> 7, c = i & 127;
  float cx = pos_a[((long)a * TT + TT - 1) * 2];
  float cy = pos_a[((long)a * TT + TT - 1) * 2 + 1];
  float f0 = (sums[0] - cx) / (float)(AA - 1);
  float f1 = (sums[1] - cy) / (float)(AA - 1);
  float f2 = sums[2] / (float)MM;
  float f3 = sums[3] / (float)MM;
  pe[i] = f0 * peW[c] + f1 * peW[128 + c] + f2 * peW[256 + c] + f3 * peW[384 + c] + peb[c];
}
__global__ void k_addpe(float* __restrict__ x, const float* __restrict__ pe) {
  long i = (long)blockIdx.x * blockDim.x + threadIdx.x;
  if (i >= (long)NTOKENS * HH) return;
  int n = (int)(i >> 7);
  int a = n / TT;
  x[i] += pe[a * HH + (int)(i & 127)];
}

// ------------------------------ LayerNorm ----------------------------------
__global__ void __launch_bounds__(128) k_layernorm(const float* __restrict__ x,
                                                   const float* __restrict__ g,
                                                   const float* __restrict__ b,
                                                   float* __restrict__ yf,
                                                   __bf16* __restrict__ ybf, int N) {
  int row = blockIdx.x * 4 + (threadIdx.x >> 5);
  int lane = threadIdx.x & 31;
  if (row >= N) return;
  const float* xr = x + (long)row * HH;
  float v0[4], s = 0.f, s2 = 0.f;
#pragma unroll
  for (int i = 0; i < 4; ++i) {
    float t = xr[lane * 4 + i];
    v0[i] = t; s += t; s2 += t * t;
  }
#pragma unroll
  for (int off = 16; off; off >>= 1) {
    s += __shfl_xor(s, off, 32);
    s2 += __shfl_xor(s2, off, 32);
  }
  float mu = s / 128.f;
  float var = s2 / 128.f - mu * mu;
  float rs = rsqrtf(var + 1e-5f);
#pragma unroll
  for (int i = 0; i < 4; ++i) {
    int c = lane * 4 + i;
    float y = (v0[i] - mu) * rs * g[c] + b[c];
    if (yf) yf[(long)row * HH + c] = y;
    if (ybf) ybf[(long)row * HH + c] = (__bf16)y;
  }
}

// ------------------------- WMMA GEMM (bf16 -> f32) -------------------------
// C[N,NCOL] = act( A[N,K] * W[K,NCOL] + bias (+ resid) ).
// RT*16 rows/block, 4 waves each owning NCOL/4 columns; B fragment reused
// across RT row tiles. A tile staged to LDS via TDM when available.
template <int NCOL, int RT>
__global__ void __launch_bounds__(128) k_gemm(const __bf16* __restrict__ Am,
                                              const __bf16* __restrict__ Wm,
                                              const float* __restrict__ bias,
                                              const float* __restrict__ resid,
                                              float* __restrict__ Cf,
                                              __bf16* __restrict__ Cbf,
                                              int N, int K, int act) {
  __shared__ __bf16 As[RT * 16 * 32];
  __shared__ __bf16 BsT[NCOL * 32];
  const int tid = threadIdx.x, lane = tid & 31, wave = tid >> 5;
  const int row0 = blockIdx.x * (RT * 16);
  const int cpw = NCOL / 4;
  const int NT = cpw / 16;
  const int colbase = wave * cpw;
  v8f acc[RT][NT];
#pragma unroll
  for (int rt = 0; rt < RT; ++rt)
#pragma unroll
    for (int nt = 0; nt < NT; ++nt)
#pragma unroll
      for (int j = 0; j < 8; ++j) acc[rt][nt][j] = 0.f;

  for (int k0 = 0; k0 < K; k0 += 32) {
    __syncthreads();
#if HAVE_TDM
    if (wave == 0) {
      // 2-D TDM descriptor: load RT*16 x 32 bf16 tile of A into LDS.
      unsigned long long ga =
          (unsigned long long)(const void*)Am + (((unsigned long long)row0 * (unsigned)K + (unsigned)k0) << 1);
      unsigned ldsa = (unsigned)(unsigned long long)(void*)As;
      unsigned td0 = (unsigned)(K - k0);                       // elems left in a row
      unsigned td1 = (unsigned)((N > row0) ? (N - row0) : 0);  // rows left (OOB -> 0)
      v4u g0;
      g0[0] = 1u;                                   // count=1, user descriptor
      g0[1] = ldsa;                                 // lds_addr
      g0[2] = (unsigned)ga;                         // global_addr[31:0]
      g0[3] = ((unsigned)(ga >> 32) & 0x01FFFFFFu) | 0x80000000u;  // addr[56:32] | type=2
      v8i g1;
      g1[0] = 0x00010000;                                        // data_size = 1 -> 2 bytes
      g1[1] = (int)((td0 & 0xFFFFu) << 16);                      // tensor_dim0 lo
      g1[2] = (int)((td0 >> 16) | ((td1 & 0xFFFFu) << 16));      // dim0 hi | dim1 lo
      g1[3] = (int)((td1 >> 16) | (32u << 16));                  // dim1 hi | tile_dim0=32
      g1[4] = (int)(RT * 16);                                    // tile_dim1 (tile_dim2=0)
      g1[5] = (int)(unsigned)K;                                  // tensor_dim0_stride lo
      g1[6] = 0;
      g1[7] = 0;
      v4i z; z[0] = 0; z[1] = 0; z[2] = 0; z[3] = 0;
      v8i z8; z8[0] = 0; z8[1] = 0; z8[2] = 0; z8[3] = 0; z8[4] = 0; z8[5] = 0; z8[6] = 0; z8[7] = 0;
      __builtin_amdgcn_tensor_load_to_lds(g0, g1, z, z, z8, 0);
      __builtin_amdgcn_s_wait_tensorcnt(0);
    }
#else
    for (int i = tid * 8; i < RT * 16 * 32; i += 128 * 8) {
      int r = i >> 5, c = i & 31;
      int grow = row0 + r;
#pragma unroll
      for (int j = 0; j < 8; ++j)
        As[i + j] = (grow < N) ? Am[(long)grow * K + k0 + c + j] : (__bf16)0.f;
    }
#endif
    for (int i = tid * 8; i < 32 * NCOL; i += 128 * 8) {
      int r = i / NCOL, c = i - r * NCOL;
#pragma unroll
      for (int j = 0; j < 8; ++j) BsT[(c + j) * 32 + r] = Wm[(long)(k0 + r) * NCOL + c + j];
    }
    if (k0 + 32 < K) __builtin_prefetch(&Wm[(long)(k0 + 32) * NCOL], 0, 1);
    __syncthreads();
    v16bf afr[RT];
#pragma unroll
    for (int rt = 0; rt < RT; ++rt) afr[rt] = load_a_frag(As + rt * 16 * 32, 32, lane);
#pragma unroll
    for (int nt = 0; nt < NT; ++nt) {
      v16bf bfr = load_bT_frag(BsT + (colbase + nt * 16) * 32, lane);
#pragma unroll
      for (int rt = 0; rt < RT; ++rt)
        acc[rt][nt] = __builtin_amdgcn_wmma_f32_16x16x32_bf16(false, afr[rt], false, bfr,
                                                              (short)0, acc[rt][nt], false, false);
    }
  }
  const int hi = lane >> 4, nn = lane & 15;
#pragma unroll
  for (int rt = 0; rt < RT; ++rt) {
#pragma unroll
    for (int nt = 0; nt < NT; ++nt) {
#pragma unroll
      for (int v = 0; v < 8; ++v) {
        int r = (hi << 3) + v;
        int c = colbase + nt * 16 + nn;
        int grow = row0 + rt * 16 + r;
        if (grow < N) {
          float val = acc[rt][nt][v];
          if (bias) val += bias[c];
          if (resid) val += resid[(long)grow * NCOL + c];
          if (act == 1) val = val > 0.f ? val : 0.f;
          else if (act == 2) val = 1.f / (1.f + __expf(-val));
          if (Cf) Cf[(long)grow * NCOL + c] = val;
          if (Cbf) Cbf[(long)grow * NCOL + c] = (__bf16)val;
        }
      }
    }
  }
}

// ------------------- fused Fourier embedding MLP (WMMA) --------------------
__global__ void __launch_bounds__(128) k_fourier(
    const float* __restrict__ x, const float* __restrict__ freqs,
    const __bf16* __restrict__ W1, const float* __restrict__ b1,
    const float* __restrict__ ln1g, const float* __restrict__ ln1b,
    const __bf16* __restrict__ W2, const float* __restrict__ b2,
    const float* __restrict__ olg, const float* __restrict__ olb,
    const __bf16* __restrict__ Wo, const float* __restrict__ bo,
    const float* __restrict__ type_emb, const int* __restrict__ agent_type,
    float* __restrict__ out, int N, int din, int catMode) {
  __shared__ __bf16 feats[16 * 160];
  __shared__ float htile[16 * 128];
  __shared__ __bf16 hbf[16 * 128];
  __shared__ float stats[32];
  __shared__ __bf16 BsT[128 * 32];
  const int tid = threadIdx.x, lane = tid & 31, wave = tid >> 5;
  const int row0 = blockIdx.x * 16;
  const int colbase = wave * 32;
  const int hi = lane >> 4, nn = lane & 15;
  v8f acc2[2];
#pragma unroll
  for (int i = 0; i < 2; ++i)
#pragma unroll
    for (int j = 0; j < 8; ++j) acc2[i][j] = 0.f;

  for (int d = 0; d < din; ++d) {
    __syncthreads();
    for (int i = tid; i < 16 * 160; i += 128) {
      int r = i / 160, f = i - r * 160;
      int grow = row0 + r;
      float xv = (grow < N) ? x[(long)grow * din + d] : 0.f;
      float val;
      if (f < 64) val = cosf(xv * freqs[d * 64 + f] * TWO_PI_F);
      else if (f < 128) val = sinf(xv * freqs[d * 64 + (f - 64)] * TWO_PI_F);
      else if (f == 128) val = xv;
      else val = 0.f;
      feats[i] = (__bf16)val;
    }
    __syncthreads();
    v8f acc1[2];
#pragma unroll
    for (int i = 0; i < 2; ++i)
#pragma unroll
      for (int j = 0; j < 8; ++j) acc1[i][j] = 0.f;
    for (int k0 = 0; k0 < 160; k0 += 32) {
      stage_bT128(W1 + (long)(d * 160 + k0) * 128, BsT, tid);
      __syncthreads();
      v16bf afr = load_a_frag(feats + k0, 160, lane);
#pragma unroll
      for (int nt = 0; nt < 2; ++nt) {
        v16bf bfr = load_bT_frag(BsT + (colbase + nt * 16) * 32, lane);
        acc1[nt] = __builtin_amdgcn_wmma_f32_16x16x32_bf16(false, afr, false, bfr,
                                                           (short)0, acc1[nt], false, false);
      }
      __syncthreads();
    }
#pragma unroll
    for (int nt = 0; nt < 2; ++nt)
#pragma unroll
      for (int v = 0; v < 8; ++v) {
        int r = (hi << 3) + v, c = colbase + nt * 16 + nn;
        htile[r * 128 + c] = acc1[nt][v] + b1[d * 128 + c];
      }
    __syncthreads();
    if (tid < 16) {
      float s = 0.f, s2 = 0.f;
      for (int c = 0; c < 128; ++c) { float t = htile[tid * 128 + c]; s += t; s2 += t * t; }
      float mu = s / 128.f, var = s2 / 128.f - mu * mu;
      stats[tid * 2] = mu;
      stats[tid * 2 + 1] = rsqrtf(var + 1e-5f);
    }
    __syncthreads();
    for (int i = tid; i < 16 * 128; i += 128) {
      int r = i >> 7, c = i & 127;
      float t = (htile[i] - stats[r * 2]) * stats[r * 2 + 1] * ln1g[d * 128 + c] + ln1b[d * 128 + c];
      hbf[i] = (__bf16)(t > 0.f ? t : 0.f);
    }
    __syncthreads();
    for (int k0 = 0; k0 < 128; k0 += 32) {
      stage_bT128(W2 + (long)(d * 128 + k0) * 128, BsT, tid);
      __syncthreads();
      v16bf afr = load_a_frag(hbf + k0, 128, lane);
#pragma unroll
      for (int nt = 0; nt < 2; ++nt) {
        v16bf bfr = load_bT_frag(BsT + (colbase + nt * 16) * 32, lane);
        acc2[nt] = __builtin_amdgcn_wmma_f32_16x16x32_bf16(false, afr, false, bfr,
                                                           (short)0, acc2[nt], false, false);
      }
      __syncthreads();
    }
#pragma unroll
    for (int nt = 0; nt < 2; ++nt)
#pragma unroll
      for (int v = 0; v < 8; ++v) acc2[nt][v] += b2[d * 128 + colbase + nt * 16 + nn];
  }
  __syncthreads();
#pragma unroll
  for (int nt = 0; nt < 2; ++nt)
#pragma unroll
    for (int v = 0; v < 8; ++v) {
      int r = (hi << 3) + v, c = colbase + nt * 16 + nn;
      int grow = row0 + r;
      float val = acc2[nt][v];
      if (catMode && grow < N) val += type_emb[agent_type[grow / TT] * 128 + c];
      htile[r * 128 + c] = val;
    }
  __syncthreads();
  if (tid < 16) {
    float s = 0.f, s2 = 0.f;
    for (int c = 0; c < 128; ++c) { float t = htile[tid * 128 + c]; s += t; s2 += t * t; }
    float mu = s / 128.f, var = s2 / 128.f - mu * mu;
    stats[tid * 2] = mu;
    stats[tid * 2 + 1] = rsqrtf(var + 1e-5f);
  }
  __syncthreads();
  for (int i = tid; i < 16 * 128; i += 128) {
    int r = i >> 7, c = i & 127;
    float t = (htile[i] - stats[r * 2]) * stats[r * 2 + 1] * olg[c] + olb[c];
    hbf[i] = (__bf16)(t > 0.f ? t : 0.f);
  }
  __syncthreads();
  v8f acc3[2];
#pragma unroll
  for (int i = 0; i < 2; ++i)
#pragma unroll
    for (int j = 0; j < 8; ++j) acc3[i][j] = 0.f;
  for (int k0 = 0; k0 < 128; k0 += 32) {
    stage_bT128(Wo + (long)k0 * 128, BsT, tid);
    __syncthreads();
    v16bf afr = load_a_frag(hbf + k0, 128, lane);
#pragma unroll
    for (int nt = 0; nt < 2; ++nt) {
      v16bf bfr = load_bT_frag(BsT + (colbase + nt * 16) * 32, lane);
      acc3[nt] = __builtin_amdgcn_wmma_f32_16x16x32_bf16(false, afr, false, bfr,
                                                         (short)0, acc3[nt], false, false);
    }
    __syncthreads();
  }
#pragma unroll
  for (int nt = 0; nt < 2; ++nt)
#pragma unroll
    for (int v = 0; v < 8; ++v) {
      int r = (hi << 3) + v, c = colbase + nt * 16 + nn;
      int grow = row0 + r;
      if (grow < N) out[(long)grow * 128 + c] = acc3[nt][v] + bo[c];
    }
}

// ------------------------ edge attention kernels ---------------------------
__global__ void k_attn_init(unsigned* __restrict__ mkey, float* __restrict__ ssum,
                            float* __restrict__ msg) {
  long i = (long)blockIdx.x * blockDim.x + threadIdx.x;
  if (i >= (long)NTOKENS * HH) return;
  msg[i] = 0.f;
  int c = (int)(i & 127);
  if (c < 8) {
    long t = (i >> 7) * 8 + c;
    mkey[t] = 0u;
    ssum[t] = 0.f;
  }
}
__global__ void k_attn_sim(const float* __restrict__ q, const float* __restrict__ k,
                           const float* __restrict__ kr, const int* __restrict__ ei, int E,
                           float* __restrict__ simb, unsigned* __restrict__ mkey) {
  long idx = (long)blockIdx.x * blockDim.x + threadIdx.x;
  if (idx >= (long)E * 8) return;
  int e = (int)(idx >> 3), h = (int)(idx & 7);
  int s = ei[e], d = ei[E + e];
  const float* qp = q + (long)d * HH + h * 16;
  const float* kp = k + (long)s * HH + h * 16;
  const float* rp = kr + (long)e * HH + h * 16;
  float acc = 0.f;
#pragma unroll
  for (int t = 0; t < 16; ++t) acc += qp[t] * (kp[t] + rp[t]);
  acc *= 0.25f;
  simb[idx] = acc;
  atomicMax(&mkey[(long)d * 8 + h], fkey(acc));
}
__global__ void k_attn_exp(float* __restrict__ simb, const unsigned* __restrict__ mkey,
                           float* __restrict__ ssum, const int* __restrict__ ei, int E) {
  long idx = (long)blockIdx.x * blockDim.x + threadIdx.x;
  if (idx >= (long)E * 8) return;
  int e = (int)(idx >> 3), h = (int)(idx & 7);
  int d = ei[E + e];
  float m = fdec(mkey[(long)d * 8 + h]);
  float w = __expf(simb[idx] - m);
  simb[idx] = w;
  atomicAdd(&ssum[(long)d * 8 + h], w);
}
__global__ void k_attn_msg(const float* __restrict__ simb, const float* __restrict__ v,
                           const float* __restrict__ vr, const int* __restrict__ ei, int E,
                           float* __restrict__ msg) {
  long idx = (long)blockIdx.x * blockDim.x + threadIdx.x;
  if (idx >= (long)E * 8) return;
  int e = (int)(idx >> 3), h = (int)(idx & 7);
  int s = ei[e], d = ei[E + e];
  float w = simb[idx];
  const float* vp = v + (long)s * HH + h * 16;
  const float* rp = vr + (long)e * HH + h * 16;
  float* mp = msg + (long)d * HH + h * 16;
#pragma unroll
  for (int t = 0; t < 16; ++t) atomicAdd(&mp[t], w * (vp[t] + rp[t]));
}
__global__ void k_msg_cat(float* __restrict__ msg, const float* __restrict__ ssum,
                          const __bf16* __restrict__ xd, __bf16* __restrict__ catb) {
  long i = (long)blockIdx.x * blockDim.x + threadIdx.x;
  if (i >= (long)NTOKENS * HH) return;
  int r = (int)(i >> 7), c = (int)(i & 127);
  float mm = msg[i] / (ssum[(long)r * 8 + (c >> 4)] + 1e-16f);
  msg[i] = mm;
  catb[(long)r * 256 + c] = (__bf16)mm;
  catb[(long)r * 256 + 128 + c] = xd[i];
}
__global__ void k_agg(const float* __restrict__ msg, const float* __restrict__ g,
                      const float* __restrict__ sp, __bf16* __restrict__ agg) {
  long i = (long)blockIdx.x * blockDim.x + threadIdx.x;
  if (i >= (long)NTOKENS * HH) return;
  float m = msg[i];
  agg[i] = (__bf16)(m + g[i] * (sp[i] - m));
}

// ============================= host launcher ===============================
extern "C" void kernel_launch(void* const* d_in, const int* in_sizes, int n_in,
                              void* d_out, int out_size, void* d_ws, size_t ws_size,
                              hipStream_t stream) {
  (void)n_in; (void)out_size; (void)ws_size;
  int cur = 0;
  auto F32 = [&]() { return (const float*)d_in[cur++]; };

  const float* pos_a = F32();
  const float* head_a = F32();
  const float* vel = F32();
  const float* pos_pl = F32();
  const float* orient_pl = F32();
  const float* x_pl = F32();
  const float* type_emb = F32();
  const float* pe_W = F32();
  const float* pe_b = F32();
  struct FP { const float *freqs,*W1,*b1,*ln1g,*ln1b,*W2,*b2,*olg,*olb,*Wo,*bo; };
  auto getFP = [&]() {
    FP p; p.freqs=F32(); p.W1=F32(); p.b1=F32(); p.ln1g=F32(); p.ln1b=F32();
    p.W2=F32(); p.b2=F32(); p.olg=F32(); p.olb=F32(); p.Wo=F32(); p.bo=F32(); return p;
  };
  FP xaF = getFP(), rtF = getFP(), rpF = getFP(), raF = getFP();
  struct AP { const float *lsg,*lsb,*ldg,*ldb,*lrg,*lrb,*Wq,*bq,*Wk,*Wv,*Wkr,*Wvr,*Ws,*bs,
                         *Wg,*bg,*Wout,*bout,*ffg,*ffb,*fW1,*fb1,*fW2,*fb2; };
  auto getAP = [&]() {
    AP p; p.lsg=F32(); p.lsb=F32(); p.ldg=F32(); p.ldb=F32(); p.lrg=F32(); p.lrb=F32();
    p.Wq=F32(); p.bq=F32(); p.Wk=F32(); p.Wv=F32(); p.Wkr=F32(); p.Wvr=F32();
    p.Ws=F32(); p.bs=F32(); p.Wg=F32(); p.bg=F32(); p.Wout=F32(); p.bout=F32();
    p.ffg=F32(); p.ffb=F32(); p.fW1=F32(); p.fb1=F32(); p.fW2=F32(); p.fb2=F32(); return p;
  };
  AP t_at[2], p_at[2], a_at[2];
  for (int l = 0; l < 2; ++l) t_at[l] = getAP();
  for (int l = 0; l < 2; ++l) p_at[l] = getAP();
  for (int l = 0; l < 2; ++l) a_at[l] = getAP();
  const int* agent_type = (const int*)d_in[cur]; cur++;
  int Et = in_sizes[cur] / 2; const int* ei_t = (const int*)d_in[cur]; cur++;
  int Ep = in_sizes[cur] / 2; const int* ei_p = (const int*)d_in[cur]; cur++;
  int Ea = in_sizes[cur] / 2; const int* ei_a = (const int*)d_in[cur]; cur++;
  int maxE = Et > Ep ? Et : Ep; if (Ea > maxE) maxE = Ea;

  char* wp = (char*)d_ws;
  auto alloc = [&](size_t bytes) -> void* {
    void* p = (void*)wp;
    wp += (bytes + 255) & ~(size_t)255;
    return p;
  };
  auto blk = [](long n) { return dim3((unsigned)((n + 255) / 256)); };

  auto cvt = [&](const float* s, long n) {
    __bf16* d = (__bf16*)alloc((size_t)n * 2);
    k_cvt_bf16<<<blk(n), 256, 0, stream>>>(s, d, n);
    return d;
  };
  struct FPB { __bf16 *W1, *W2, *Wo; };
  auto getFPB = [&](const FP& p, int din) {
    FPB b;
    b.W1 = (__bf16*)alloc((size_t)din * 160 * 128 * 2);
    k_pad_w1<<<blk((long)din * 160 * 128), 256, 0, stream>>>(p.W1, b.W1, din);
    b.W2 = cvt(p.W2, (long)din * 128 * 128);
    b.Wo = cvt(p.Wo, 128 * 128);
    return b;
  };
  FPB xaFB = getFPB(xaF, 4), rtFB = getFPB(rtF, 4), rpFB = getFPB(rpF, 3), raFB = getFPB(raF, 3);
  struct APB { __bf16 *Wq,*Wk,*Wv,*Wkr,*Wvr,*Ws,*Wg,*Wout,*f1,*f2; };
  auto getAPB = [&](const AP& p) {
    APB b;
    b.Wq = cvt(p.Wq, 128 * 128); b.Wk = cvt(p.Wk, 128 * 128); b.Wv = cvt(p.Wv, 128 * 128);
    b.Wkr = cvt(p.Wkr, 128 * 128); b.Wvr = cvt(p.Wvr, 128 * 128); b.Ws = cvt(p.Ws, 128 * 128);
    b.Wg = cvt(p.Wg, 256 * 128); b.Wout = cvt(p.Wout, 128 * 128);
    b.f1 = cvt(p.fW1, 128 * 512); b.f2 = cvt(p.fW2, 512 * 128);
    return b;
  };
  APB t_ab[2], p_ab[2], a_ab[2];
  for (int l = 0; l < 2; ++l) t_ab[l] = getAPB(t_at[l]);
  for (int l = 0; l < 2; ++l) p_ab[l] = getAPB(p_at[l]);
  for (int l = 0; l < 2; ++l) a_ab[l] = getAPB(a_at[l]);

  float* xfeat = (float*)alloc((size_t)NTOKENS * 4 * 4);
  float* rtfeat = (float*)alloc((size_t)Et * 4 * 4);
  float* rpfeat = (float*)alloc((size_t)Ep * 3 * 4);
  float* rafeat = (float*)alloc((size_t)Ea * 3 * 4);
  float* xA = (float*)alloc((size_t)NTOKENS * HH * 4);
  float* xB = (float*)alloc((size_t)NTOKENS * HH * 4);
  float* xC = (float*)alloc((size_t)NTOKENS * HH * 4);
  float* r_t = (float*)alloc((size_t)Et * HH * 4);
  float* r_p = (float*)alloc((size_t)Ep * HH * 4);
  float* r_a = (float*)alloc((size_t)Ea * HH * 4);
  float* xpls = (float*)alloc((size_t)NPLS * HH * 4);
  float* sums = (float*)alloc(4 * 4);
  float* pe = (float*)alloc((size_t)AA * HH * 4);
  __bf16* xs_ln = (__bf16*)alloc((size_t)NPLS * HH * 2);
  __bf16* xd_ln = (__bf16*)alloc((size_t)NTOKENS * HH * 2);
  __bf16* rn = (__bf16*)alloc((size_t)maxE * HH * 2);
  float* qf = (float*)alloc((size_t)NTOKENS * HH * 4);
  float* kf = (float*)alloc((size_t)NPLS * HH * 4);
  float* vf = (float*)alloc((size_t)NPLS * HH * 4);
  float* krf = (float*)alloc((size_t)maxE * HH * 4);
  float* vrf = (float*)alloc((size_t)maxE * HH * 4);
  float* simb = (float*)alloc((size_t)maxE * 8 * 4);
  unsigned* mkey = (unsigned*)alloc((size_t)NTOKENS * 8 * 4);
  float* ssum = (float*)alloc((size_t)NTOKENS * 8 * 4);
  float* msg = (float*)alloc((size_t)NTOKENS * HH * 4);
  __bf16* catb = (__bf16*)alloc((size_t)NTOKENS * 256 * 2);
  float* gf = (float*)alloc((size_t)NTOKENS * HH * 4);
  float* sprj = (float*)alloc((size_t)NTOKENS * HH * 4);
  __bf16* aggb = (__bf16*)alloc((size_t)NTOKENS * HH * 2);
  float* xnew = (float*)alloc((size_t)NTOKENS * HH * 4);
  __bf16* ffh = (__bf16*)alloc((size_t)NTOKENS * HH * 2);
  __bf16* ff1 = (__bf16*)alloc((size_t)NTOKENS * 512 * 2);

  auto ln = [&](const float* x, const float* g, const float* b, float* yf, __bf16* ybf, int n) {
    k_layernorm<<<dim3((unsigned)((n + 3) / 4)), 128, 0, stream>>>(x, g, b, yf, ybf, n);
  };
  auto gemm = [&](const __bf16* A, const __bf16* W, const float* bias, const float* resid,
                  float* Cf, __bf16* Cbf, int n, int K, int ncol, int act) {
    if (ncol == 128) {
      dim3 g((unsigned)((n + 63) / 64));
      k_gemm<128, 4><<<g, 128, 0, stream>>>(A, W, bias, resid, Cf, Cbf, n, K, act);
    } else {
      dim3 g((unsigned)((n + 31) / 32));
      k_gemm<512, 2><<<g, 128, 0, stream>>>(A, W, bias, resid, Cf, Cbf, n, K, act);
    }
  };
  auto fourier = [&](const FP& p, const FPB& pb, const float* x, int n, int din, int catMode,
                     float* outp) {
    k_fourier<<<dim3((unsigned)((n + 15) / 16)), 128, 0, stream>>>(
        x, p.freqs, pb.W1, p.b1, p.ln1g, p.ln1b, pb.W2, p.b2, p.olg, p.olb, pb.Wo, p.bo,
        type_emb, agent_type, outp, n, din, catMode);
  };

  k_feat_agent<<<blk(NTOKENS), 256, 0, stream>>>(pos_a, head_a, vel, xfeat);
  k_feat_rt<<<blk(Et), 256, 0, stream>>>(pos_a, head_a, ei_t, Et, rtfeat);
  k_feat_rpl2a<<<blk(Ep), 256, 0, stream>>>(pos_a, head_a, pos_pl, orient_pl, ei_p, Ep, rpfeat);
  k_feat_ra2a<<<blk(Ea), 256, 0, stream>>>(pos_a, head_a, ei_a, Ea, rafeat);

  fourier(xaF, xaFB, xfeat, NTOKENS, 4, 1, xA);
  fourier(rtF, rtFB, rtfeat, Et, 4, 0, r_t);
  fourier(rpF, rpFB, rpfeat, Ep, 3, 0, r_p);
  fourier(raF, raFB, rafeat, Ea, 3, 0, r_a);

  k_reduce_pe<<<1, 256, 0, stream>>>(pos_a, pos_pl, sums);
  k_pe<<<blk((long)AA * HH), 256, 0, stream>>>(sums, pos_a, pe_W, pe_b, pe);
  k_addpe<<<blk((long)NTOKENS * HH), 256, 0, stream>>>(xA, pe);

  k_xpls<<<blk((long)MM * TT * HH), 256, 0, stream>>>(x_pl, xpls);

  auto attn = [&](const AP& p, const APB& pb, const float* xsrc, int nsrc, const float* xdst,
                  const float* r, const int* ei, int E, float* xout) {
    ln(xsrc, p.lsg, p.lsb, nullptr, xs_ln, nsrc);
    ln(xdst, p.ldg, p.ldb, nullptr, xd_ln, NTOKENS);
    ln(r, p.lrg, p.lrb, nullptr, rn, E);
    gemm(xd_ln, pb.Wq, p.bq, nullptr, qf, nullptr, NTOKENS, 128, 128, 0);
    gemm(xs_ln, pb.Wk, nullptr, nullptr, kf, nullptr, nsrc, 128, 128, 0);
    gemm(xs_ln, pb.Wv, nullptr, nullptr, vf, nullptr, nsrc, 128, 128, 0);
    gemm(rn, pb.Wkr, nullptr, nullptr, krf, nullptr, E, 128, 128, 0);
    gemm(rn, pb.Wvr, nullptr, nullptr, vrf, nullptr, E, 128, 128, 0);
    k_attn_init<<<blk((long)NTOKENS * HH), 256, 0, stream>>>(mkey, ssum, msg);
    k_attn_sim<<<blk((long)E * 8), 256, 0, stream>>>(qf, kf, krf, ei, E, simb, mkey);
    k_attn_exp<<<blk((long)E * 8), 256, 0, stream>>>(simb, mkey, ssum, ei, E);
    k_attn_msg<<<blk((long)E * 8), 256, 0, stream>>>(simb, vf, vrf, ei, E, msg);
    k_msg_cat<<<blk((long)NTOKENS * HH), 256, 0, stream>>>(msg, ssum, xd_ln, catb);
    gemm(catb, pb.Wg, p.bg, nullptr, gf, nullptr, NTOKENS, 256, 128, 2);
    gemm(xd_ln, pb.Ws, p.bs, nullptr, sprj, nullptr, NTOKENS, 128, 128, 0);
    k_agg<<<blk((long)NTOKENS * HH), 256, 0, stream>>>(msg, gf, sprj, aggb);
    gemm(aggb, pb.Wout, p.bout, xdst, xnew, nullptr, NTOKENS, 128, 128, 0);
    ln(xnew, p.ffg, p.ffb, nullptr, ffh, NTOKENS);
    gemm(ffh, pb.f1, p.fb1, nullptr, nullptr, ff1, NTOKENS, 128, 512, 1);
    gemm(ff1, pb.f2, p.fb2, xnew, xout, nullptr, NTOKENS, 512, 128, 0);
  };

  for (int l = 0; l < 2; ++l) {
    attn(t_at[l], t_ab[l], xA, NTOKENS, xA, r_t, ei_t, Et, xB);
    k_perm<<<blk((long)NTOKENS * HH), 256, 0, stream>>>(xB, xC, 1);
    attn(p_at[l], p_ab[l], xpls, NPLS, xC, r_p, ei_p, Ep, xB);
    attn(a_at[l], a_ab[l], xB, NTOKENS, xB, r_a, ei_a, Ea, xC);
    k_perm<<<blk((long)NTOKENS * HH), 256, 0, stream>>>(xC, xA, 0);
  }

  k_copy<<<blk((long)NTOKENS * HH), 256, 0, stream>>>(xA, (float*)d_out, (long)NTOKENS * HH);
}
